// ContrastiveLoss_cosine2_65008624992272
// MI455X (gfx1250) — matrix operand order repaired and checked by reading, not compiled, not run on hardware
//
#include <hip/hip_runtime.h>
#include <hip/hip_bf16.h>
#include <stddef.h>

typedef __attribute__((ext_vector_type(16))) _Float16 v16h;
typedef __attribute__((ext_vector_type(8)))  float    v8f;

#define NW 8   // waves per block in main kernel
#define NQ 5   // dot-product groups

__device__ __forceinline__ v16h cvt16(float4 a, float4 b, float4 c, float4 d) {
  v16h r;
  r[0]=(_Float16)a.x;  r[1]=(_Float16)a.y;  r[2]=(_Float16)a.z;  r[3]=(_Float16)a.w;
  r[4]=(_Float16)b.x;  r[5]=(_Float16)b.y;  r[6]=(_Float16)b.z;  r[7]=(_Float16)b.w;
  r[8]=(_Float16)c.x;  r[9]=(_Float16)c.y;  r[10]=(_Float16)c.z; r[11]=(_Float16)c.w;
  r[12]=(_Float16)d.x; r[13]=(_Float16)d.y; r[14]=(_Float16)d.z; r[15]=(_Float16)d.w;
  return r;
}

// A-matrix fragment (16x32 f16): lane m=lane&15 holds 16 contiguous K,
// lanes 0-15 -> K 0..15, lanes 16-31 -> K 16..31 (matches load_b).
__device__ __forceinline__ v16h load_a(const float* __restrict__ rowp, int kc, int aoff) {
  const float4* p = reinterpret_cast<const float4*>(rowp + kc + aoff);
  return cvt16(p[0], p[1], p[2], p[3]);
}

// B-matrix fragment (32x16 f16): lane n=lane&15 holds 16 contiguous K,
// lanes 0-15 -> K 0..15, lanes 16-31 -> K 16..31.
__device__ __forceinline__ v16h load_b(const float* __restrict__ rowp, int kc, int boff) {
  const float4* p = reinterpret_cast<const float4*>(rowp + kc + boff);
  return cvt16(p[0], p[1], p[2], p[3]);
}

// ---------------- kernel 1: per-row L2 norms ----------------
__global__ __launch_bounds__(256) void ccl_norms(const float* __restrict__ img,
                                                 const float* __restrict__ txt,
                                                 float* __restrict__ norms,
                                                 int B, int D) {
  const int r = blockIdx.x;                 // 0 .. 2B-1
  const float* row = (r < B) ? (img + (size_t)r * D) : (txt + (size_t)(r - B) * D);
  const float4* p = reinterpret_cast<const float4*>(row);
  const int n4 = D >> 2;
  float s = 0.f;
  for (int i = threadIdx.x; i < n4; i += blockDim.x) {
    float4 v = p[i];
    s += v.x * v.x + v.y * v.y + v.z * v.z + v.w * v.w;
  }
  __shared__ float sd[256];
  sd[threadIdx.x] = s;
  __syncthreads();
  for (int off = 128; off > 0; off >>= 1) {
    if (threadIdx.x < off) sd[threadIdx.x] += sd[threadIdx.x + off];
    __syncthreads();
  }
  if (threadIdx.x == 0) norms[r] = sqrtf(sd[0]);
}

// ---------------- kernel 2: WMMA batched dots + mining + loss ----------------
__global__ __launch_bounds__(256) void ccl_main(const float* __restrict__ img,
                                                const float* __restrict__ txt,
                                                const int* __restrict__ candI,
                                                const int* __restrict__ candT,
                                                const float* __restrict__ norms,
                                                float* __restrict__ out,
                                                float* __restrict__ partials,
                                                int B, int D) {
  __shared__ float s[NW][NQ][32][8];   // per-wave C fragments (40 KB)
  __shared__ float lossRow[16];

  const int tid  = threadIdx.x;
  const int w    = tid >> 5;
  const int lane = tid & 31;
  const int n    = lane & 15;
  const int tileBase = blockIdx.x * 16;
  const int gi   = tileBase + n;

  // Row pointers this lane serves (A row m == B column n == tile row n).
  const float* rI  = img + (size_t)gi * D;
  const float* rT  = txt + (size_t)gi * D;
  const float* rI0 = img + (size_t)candI[2 * gi]     * D;
  const float* rI1 = img + (size_t)candI[2 * gi + 1] * D;
  const float* rT0 = txt + (size_t)candT[2 * gi]     * D;
  const float* rT1 = txt + (size_t)candT[2 * gi + 1] * D;

  const int aoff = (lane >= 16) ? 16 : 0;
  const int boff = (lane >= 16) ? 16 : 0;

  v8f acc[NQ];
#pragma unroll
  for (int q = 0; q < NQ; ++q)
#pragma unroll
    for (int j = 0; j < 8; ++j) acc[q][j] = 0.f;

  const int kPerWave = D / NW;              // 256 for D=2048
  const int k0 = w * kPerWave;
  for (int kc = k0; kc < k0 + kPerWave; kc += 32) {
    // hint the gathered (random-row) streams toward the 192MB L2
    __builtin_prefetch(rI0 + kc + 32, 0, 0);
    __builtin_prefetch(rI1 + kc + 32, 0, 0);
    __builtin_prefetch(rT0 + kc + 32, 0, 0);
    __builtin_prefetch(rT1 + kc + 32, 0, 0);

    v16h aI  = load_a(rI,  kc, aoff);
    v16h aT  = load_a(rT,  kc, aoff);
    v16h bT  = load_b(rT,  kc, boff);
    v16h bI0 = load_b(rI0, kc, boff);
    v16h bI1 = load_b(rI1, kc, boff);
    v16h bT0 = load_b(rT0, kc, boff);
    v16h bT1 = load_b(rT1, kc, boff);

    acc[0] = __builtin_amdgcn_wmma_f32_16x16x32_f16(false, aI, false, bT,  (short)0, acc[0], false, false);
    acc[1] = __builtin_amdgcn_wmma_f32_16x16x32_f16(false, aT, false, bI0, (short)0, acc[1], false, false);
    acc[2] = __builtin_amdgcn_wmma_f32_16x16x32_f16(false, aT, false, bI1, (short)0, acc[2], false, false);
    acc[3] = __builtin_amdgcn_wmma_f32_16x16x32_f16(false, aI, false, bT0, (short)0, acc[3], false, false);
    acc[4] = __builtin_amdgcn_wmma_f32_16x16x32_f16(false, aI, false, bT1, (short)0, acc[4], false, false);
  }

  // Dump C fragments so the epilogue can pull diagonals across waves.
#pragma unroll
  for (int q = 0; q < NQ; ++q)
#pragma unroll
    for (int j = 0; j < 8; ++j) s[w][q][lane][j] = acc[q][j];
  __syncthreads();

  if (tid < 16) {
    const int m  = tid;
    const int lm = (m < 8) ? m : (m + 16);  // diag lane (ISA 7.12.2 C layout)
    const int jv = m & 7;                   // diag VGPR index
    float dot[NQ];
#pragma unroll
    for (int q = 0; q < NQ; ++q) {
      float v = 0.f;
#pragma unroll
      for (int ww = 0; ww < NW; ++ww) v += s[ww][q][lm][jv];
      dot[q] = v;
    }

    const int i = tileBase + m;
    const float EPS = 1e-8f, MARGIN = 0.2f;
    const float na = norms[i];
    const float nt = norms[B + i];
    const int ci0 = candI[2 * i], ci1 = candI[2 * i + 1];
    const int ct0 = candT[2 * i], ct1 = candT[2 * i + 1];
    const float nI0 = norms[ci0],     nI1 = norms[ci1];
    const float nT0 = norms[B + ct0], nT1 = norms[B + ct1];

    const float den = na * nt;
    const float pos = 1.f - dot[0] / fmaxf(den, EPS);
    const float cosv = dot[0] / den;

    // mine img negatives (anchor = text); tie -> later candidate
    const float dA = 1.f - dot[1] / fmaxf(nt * nI0, EPS);
    const float dB = 1.f - dot[2] / fmaxf(nt * nI1, EPS);
    const float t2i_neg = (dB <= dA) ? dB : dA;
    // mine text negatives (anchor = img)
    const float eA = 1.f - dot[3] / fmaxf(na * nT0, EPS);
    const float eB = 1.f - dot[4] / fmaxf(na * nT1, EPS);
    const float i2t_neg = (eB <= eA) ? eB : eA;

    const float li2t = fmaxf(0.f, pos - i2t_neg + MARGIN);
    const float lt2i = fmaxf(0.f, pos - t2i_neg + MARGIN);

    out[1 + i]     = cosv;   // i2t_cosine
    out[1 + B + i] = cosv;   // t2i_cosine (identical by symmetry)
    lossRow[m] = li2t + lt2i;
  }
  __syncthreads();

  if (tid == 0) {
    float p = 0.f;
#pragma unroll
    for (int m = 0; m < 16; ++m) p += lossRow[m];
    partials[blockIdx.x] = p;
  }
}

// ---------------- kernel 3: deterministic final reduction ----------------
__global__ __launch_bounds__(256) void ccl_reduce(const float* __restrict__ partials,
                                                  float* __restrict__ out,
                                                  int tiles, int B) {
  __shared__ float sd[256];
  const int t = threadIdx.x;
  float v = 0.f;
  for (int i = t; i < tiles; i += 256) v += partials[i];
  sd[t] = v;
  __syncthreads();
  for (int off = 128; off > 0; off >>= 1) {
    if (t < off) sd[t] += sd[t + off];
    __syncthreads();
  }
  if (t == 0) out[0] = sd[0] / (float)B;   // mean(i2t) + mean(t2i)
}

extern "C" void kernel_launch(void* const* d_in, const int* in_sizes, int n_in,
                              void* d_out, int out_size, void* d_ws, size_t ws_size,
                              hipStream_t stream) {
  (void)n_in; (void)out_size; (void)ws_size;
  const float* img  = (const float*)d_in[0];
  const float* txt  = (const float*)d_in[1];
  const int* candI  = (const int*)d_in[4];
  const int* candT  = (const int*)d_in[5];
  const int B = in_sizes[2];            // labels length
  const int D = in_sizes[0] / B;        // 2048

  float* out      = (float*)d_out;
  float* wsf      = (float*)d_ws;
  float* norms    = wsf;                // 2B floats
  float* partials = wsf + 2 * B;        // B/16 floats

  ccl_norms<<<2 * B, 256, 0, stream>>>(img, txt, norms, B, D);
  const int tiles = B / 16;
  ccl_main<<<tiles, 256, 0, stream>>>(img, txt, candI, candT, norms, out, partials, B, D);
  ccl_reduce<<<1, 256, 0, stream>>>(partials, out, tiles, B);
}
